// CrossAttention_noise_84061099917658
// MI455X (gfx1250) — compile-verified, matrix operand order checked by reading, not run on hardware
//
#include <hip/hip_runtime.h>
#include <hip/hip_bf16.h>
#include <math.h>
#include <stdint.h>

// ---------------------------------------------------------------------------
// CrossAttention DiT block for MI455X (gfx1250, wave32, WMMA + TDM).
// All matmul stages use v_wmma_f32_16x16x32_bf16 (bf16 in, f32 accum).
// Big GEMMs (fc / ff1 / ff2) use 128x128 C-tiles with double-buffered LDS
// staged by the Tensor Data Mover (tensor_load_to_lds, TENSORcnt).
//
// Shapes: B=4, T1=T2=1024, D=1024, H=16, DK=64, K=13, FF=4096.
// ---------------------------------------------------------------------------

typedef __attribute__((ext_vector_type(16))) __bf16 v16bf;
typedef __attribute__((ext_vector_type(8)))  __bf16 v8bf;
typedef __attribute__((ext_vector_type(8)))  float  v8f;
typedef unsigned int v4u __attribute__((ext_vector_type(4)));
typedef int          v8i __attribute__((ext_vector_type(8)));

#define DEVINL __device__ __forceinline__

static constexpr int BB  = 4;
static constexpr int TT  = 1024;
static constexpr int DD  = 1024;
static constexpr int HH  = 16;
static constexpr int DKh = 64;
static constexpr int KW  = 13;
static constexpr int FFd = 4096;
static constexpr int SIXD = 6 * DD;

// ---------------------------------------------------------------------------
// WMMA helpers.
//
// Fragment convention (wave32, CDNA5 ISA 7.12.2):
//  A (16x32 MxK): lane l holds row m=l&15; halves 0..7 = K [8*(l>>4) .. +7],
//                 halves 8..15 = K [16+8*(l>>4) .. +7]  -> two 16B chunks.
//  B (32x16 KxN) for C = X @ W^T: B[k,n] = W[n,k] -> identical load pattern
//                 from row-major W (lane = n row).
//  C/D (16x16 f32): acc[r] = C[m = r + 8*(l>>4), n = l&15].
// ---------------------------------------------------------------------------
DEVINL v16bf load_frag_rm(const __bf16* base, int ld) {
    int lane = threadIdx.x & 31;
    const __bf16* p = base + (lane & 15) * ld + ((lane >> 4) << 3);
    v8bf lo = *(const v8bf*)p;
    v8bf hi = *(const v8bf*)(p + 16);
    v16bf f;
#pragma unroll
    for (int i = 0; i < 8; ++i) { f[i] = lo[i]; f[i + 8] = hi[i]; }
    return f;
}

DEVINL v8f wmma_bf16(v16bf a, v16bf b, v8f c) {
    return __builtin_amdgcn_wmma_f32_16x16x32_bf16(
        /*neg_a=*/false, a, /*neg_b=*/false, b,
        /*c_mod=*/(short)0, c, /*reuse_a=*/false, /*reuse_b=*/false);
}

DEVINL float gelu_tanh(float x) {
    float x3 = x * x * x;
    return 0.5f * x * (1.0f + tanhf(0.7978845608028654f * (x + 0.044715f * x3)));
}

// ---------------------------------------------------------------------------
// Tensor Data Mover: async-load a 2D tile (tile_rows x tile_k bf16 elements)
// from a row-major (rows_total x kd_elems) tensor into LDS, row-major stride
// tile_k.  D# built per CDNA5 ISA 8.3/8.4; 2-group form (<=2D tensor).
// Tracked with TENSORcnt; pair with __builtin_amdgcn_s_wait_tensorcnt.
// ---------------------------------------------------------------------------
DEVINL void tdm_load_2d(const __bf16* gptr, void* lptr,
                        int kd_elems, int rows_total,
                        int tile_rows, int tile_k) {
    unsigned long long ga = (unsigned long long)(uintptr_t)gptr;   // byte addr
    unsigned int lds = (unsigned int)(uintptr_t)lptr;              // LDS offset
    v4u g0;
    g0[0] = 1u;                                        // count=1, user D#
    g0[1] = lds;                                       // lds_addr [63:32]
    g0[2] = (unsigned int)(ga & 0xFFFFFFFFu);          // global_addr[31:0]
    g0[3] = (unsigned int)((ga >> 32) & 0x01FFFFFFu)   // global_addr[56:32]
          | (2u << 30);                                // type=2 ("image")
    v8i g1;
    g1[0] = (1 << 16);                                 // data_size=1 (2B), mask=0
    g1[1] = (int)((unsigned)(kd_elems & 0xFFFF) << 16);        // tensor_dim0 lo
    g1[2] = (int)(((unsigned)kd_elems >> 16) & 0xFFFFu)        // tensor_dim0 hi
          | (int)((unsigned)(rows_total & 0xFFFF) << 16);      // tensor_dim1 lo
    g1[3] = (int)(((unsigned)rows_total >> 16) & 0xFFFFu)      // tensor_dim1 hi
          | (int)((unsigned)(tile_k & 0xFFFF) << 16);          // tile_dim0
    g1[4] = (int)(tile_rows & 0xFFFF);                 // tile_dim1 (tile_dim2=0)
    g1[5] = kd_elems;                                  // tensor_dim0_stride[31:0]
    g1[6] = 0;
    g1[7] = 0;
    asm volatile("tensor_load_to_lds %0, %1" :: "s"(g0), "s"(g1) : "memory");
}

// ---------------------------------------------------------------------------
// fp32 -> bf16 conversion (weights + one-off tensors)
// ---------------------------------------------------------------------------
__global__ void f32_to_bf16_kernel(const float* __restrict__ src,
                                   __bf16* __restrict__ dst, size_t n) {
    size_t i = (size_t)blockIdx.x * blockDim.x + threadIdx.x;
    size_t st = (size_t)gridDim.x * blockDim.x;
    for (; i < n; i += st) dst[i] = (__bf16)src[i];
}

// ---------------------------------------------------------------------------
// Affine LayerNorm over rows of length D=1024, eps param.  Optional f32
// and/or bf16 outputs.  One 256-thread block per row.
// ---------------------------------------------------------------------------
__global__ void ln_rows_kernel(const float* __restrict__ x,
                               const float* __restrict__ g,
                               const float* __restrict__ bt,
                               float eps,
                               float* __restrict__ y32,
                               __bf16* __restrict__ y16) {
    __shared__ float red[256];
    int row = blockIdx.x, tid = threadIdx.x;
    const float* xr = x + (size_t)row * DD;
    float s = 0.f;
    for (int c = tid; c < DD; c += 256) s += xr[c];
    red[tid] = s; __syncthreads();
    for (int o = 128; o > 0; o >>= 1) { if (tid < o) red[tid] += red[tid + o]; __syncthreads(); }
    float mu = red[0] * (1.0f / DD);
    __syncthreads();
    float vs = 0.f;
    for (int c = tid; c < DD; c += 256) { float d = xr[c] - mu; vs += d * d; }
    red[tid] = vs; __syncthreads();
    for (int o = 128; o > 0; o >>= 1) { if (tid < o) red[tid] += red[tid + o]; __syncthreads(); }
    float rstd = rsqrtf(red[0] * (1.0f / DD) + eps);
    for (int c = tid; c < DD; c += 256) {
        float v = (xr[c] - mu) * rstd * g[c] + bt[c];
        if (y32) y32[(size_t)row * DD + c] = v;
        if (y16) y16[(size_t)row * DD + c] = (__bf16)v;
    }
}

// ---------------------------------------------------------------------------
// Non-affine LN (eps=1e-6) + AdaLN modulation: y = ln(x)*(1+scale)+shift.
// ---------------------------------------------------------------------------
__global__ void ln_mod_kernel(const float* __restrict__ x,
                              const float* __restrict__ emb,
                              int shift_off, int scale_off,
                              __bf16* __restrict__ y16) {
    __shared__ float red[256];
    int row = blockIdx.x, tid = threadIdx.x;
    int b = row / TT;
    const float* xr = x + (size_t)row * DD;
    float s = 0.f;
    for (int c = tid; c < DD; c += 256) s += xr[c];
    red[tid] = s; __syncthreads();
    for (int o = 128; o > 0; o >>= 1) { if (tid < o) red[tid] += red[tid + o]; __syncthreads(); }
    float mu = red[0] * (1.0f / DD);
    __syncthreads();
    float vs = 0.f;
    for (int c = tid; c < DD; c += 256) { float d = xr[c] - mu; vs += d * d; }
    red[tid] = vs; __syncthreads();
    for (int o = 128; o > 0; o >>= 1) { if (tid < o) red[tid] += red[tid + o]; __syncthreads(); }
    float rstd = rsqrtf(red[0] * (1.0f / DD) + 1e-6f);
    const float* eb = emb + (size_t)b * SIXD;
    for (int c = tid; c < DD; c += 256) {
        float v = (xr[c] - mu) * rstd;
        y16[(size_t)row * DD + c] = (__bf16)(v * (1.0f + eb[scale_off + c]) + eb[shift_off + c]);
    }
}

// ---------------------------------------------------------------------------
// silu(t) into zero-padded 16 x 1024 bf16 (rows >= B zero) for full WMMA tiles
// ---------------------------------------------------------------------------
__global__ void silu_pad_kernel(const float* __restrict__ t, __bf16* __restrict__ st) {
    int i = blockIdx.x * blockDim.x + threadIdx.x;   // 16*1024 threads
    int r = i >> 10;
    float v = 0.f;
    if (r < BB) { float x = t[i]; v = x / (1.0f + __expf(-x)); }
    st[i] = (__bf16)v;
}

// ---------------------------------------------------------------------------
// emb = silu(t) @ ada_w^T + ada_b.   M=16(pad of 4), N=6144, K=1024.
// ---------------------------------------------------------------------------
__global__ void ada_gemm_kernel(const __bf16* __restrict__ st,
                                const __bf16* __restrict__ wbf,
                                const float* __restrict__ bias,
                                float* __restrict__ emb) {
    int ntile = blockIdx.x;
    int lane = threadIdx.x & 31;
    const __bf16* Bb = wbf + (size_t)(ntile * 16) * DD;
    v8f acc = {};
#pragma unroll 4
    for (int ks = 0; ks < DD / 32; ++ks) {
        v16bf a = load_frag_rm(st + ks * 32, DD);
        v16bf b = load_frag_rm(Bb + ks * 32, DD);
        acc = wmma_bf16(a, b, acc);
    }
    int n = ntile * 16 + (lane & 15);
    int mlo = (lane >> 4) * 8;
    float bo = bias[n];
#pragma unroll
    for (int r = 0; r < 8; ++r) {
        int m = mlo + r;
        if (m < BB) emb[(size_t)m * SIXD + n] = acc[r] + bo;
    }
}

// ---------------------------------------------------------------------------
// Grouped conv1d (H=16 groups of 64ch, K=13, same padding) as 13 shifted
// 64x64 GEMMs per head, all WMMA.  Grid: B*H*(T/64), block 256.
// ---------------------------------------------------------------------------
__global__ void conv_qkv_kernel(const __bf16* __restrict__ xin,
                                const __bf16* __restrict__ wbf,   // (1024,64,13)
                                const float* __restrict__ bias,
                                __bf16* __restrict__ out) {
    __shared__ __bf16 Xl[76 * 64];   // 64 rows + 12 halo
    __shared__ __bf16 Wl[64 * 64];
    int qb = blockIdx.x & 15;
    int h  = (blockIdx.x >> 4) & 15;
    int b  = blockIdx.x >> 8;
    int tid = threadIdx.x, wid = tid >> 5, lane = tid & 31;

    for (int idx = tid; idx < 76 * 64; idx += 256) {
        int r = idx >> 6, c = idx & 63;
        int t = qb * 64 + r - 6;
        __bf16 v = (__bf16)0.f;
        if (t >= 0 && t < TT)
            v = xin[((size_t)(b * TT + t)) * DD + h * DKh + c];
        Xl[idx] = v;
    }
    int tm0 = wid >> 2, tn0 = wid & 3;   // tiles wid and wid+8 -> tm0, tm0+2, same tn
    int tm1 = tm0 + 2;
    v8f acc0 = {}, acc1 = {};

    for (int kap = 0; kap < KW; ++kap) {
        __syncthreads();
        for (int idx = tid; idx < 4096; idx += 256) {
            int o = idx >> 6, i = idx & 63;
            Wl[idx] = wbf[((size_t)((h * DKh + o) * DKh + i)) * KW + kap];
        }
        __syncthreads();
#pragma unroll
        for (int ks = 0; ks < 2; ++ks) {
            v16bf bfr = load_frag_rm(&Wl[(tn0 * 16) * 64 + ks * 32], 64);
            v16bf a0  = load_frag_rm(&Xl[(tm0 * 16 + kap) * 64 + ks * 32], 64);
            acc0 = wmma_bf16(a0, bfr, acc0);
            v16bf a1  = load_frag_rm(&Xl[(tm1 * 16 + kap) * 64 + ks * 32], 64);
            acc1 = wmma_bf16(a1, bfr, acc1);
        }
    }
    int n = tn0 * 16 + (lane & 15);
    int mlo = (lane >> 4) * 8;
    float bo = bias[h * DKh + n];
#pragma unroll
    for (int r = 0; r < 8; ++r) {
        int t0 = qb * 64 + tm0 * 16 + mlo + r;
        out[((size_t)(b * TT + t0)) * DD + h * DKh + n] = (__bf16)(acc0[r] + bo);
        int t1 = qb * 64 + tm1 * 16 + mlo + r;
        out[((size_t)(b * TT + t1)) * DD + h * DKh + n] = (__bf16)(acc1[r] + bo);
    }
}

// ---------------------------------------------------------------------------
// Flash cross-attention per (b, head, 64-query block).  Key mask is uniform
// per batch, so the key loop runs only to ceil(len/64); the tail block masks
// j >= len.  QK^T and PV on WMMA; online softmax in f32.
// ---------------------------------------------------------------------------
__global__ void attn_kernel(const __bf16* __restrict__ qbuf,
                            const __bf16* __restrict__ kbuf,
                            const __bf16* __restrict__ vbuf,
                            const int* __restrict__ clean_len,
                            __bf16* __restrict__ out) {
    __shared__ __bf16 Ql[64 * 64], Kl[64 * 64], VTl[64 * 64], Pl[64 * 64];
    __shared__ float  Sl[64 * 64];
    __shared__ float  mrow[64], lrow[64], arow[64];
    int qb = blockIdx.x & 15;
    int h  = (blockIdx.x >> 4) & 15;
    int b  = blockIdx.x >> 8;
    int tid = threadIdx.x, wid = tid >> 5, lane = tid & 31;
    int len = clean_len[b];
    if (len > TT) len = TT;
    if (len < 0) len = 0;

    for (int idx = tid; idx < 4096; idx += 256) {
        int r = idx >> 6, c = idx & 63;
        Ql[idx] = qbuf[((size_t)(b * TT + qb * 64 + r)) * DD + h * DKh + c];
    }
    if (tid < 64) { mrow[tid] = -3.0e38f; lrow[tid] = 0.f; }

    int tm0 = wid >> 2, tn0 = wid & 3, tm1 = tm0 + 2;
    v8f o0 = {}, o1 = {};
    int nn = tn0 * 16 + (lane & 15);
    int mlo = (lane >> 4) * 8;

    int nkb = (len + 63) >> 6;
    for (int kb = 0; kb < nkb; ++kb) {
        __syncthreads();
        for (int idx = tid; idx < 4096; idx += 256) {
            int r = idx >> 6, c = idx & 63;
            size_t src = ((size_t)(b * TT + kb * 64 + r)) * DD + h * DKh + c;
            Kl[idx] = kbuf[src];
            VTl[c * 64 + r] = vbuf[src];   // stage V transposed: VT[d][j]
        }
        __syncthreads();

        v8f s0 = {}, s1 = {};
#pragma unroll
        for (int ks = 0; ks < 2; ++ks) {
            v16bf bfr = load_frag_rm(&Kl[(tn0 * 16) * 64 + ks * 32], 64);
            v16bf a0  = load_frag_rm(&Ql[(tm0 * 16) * 64 + ks * 32], 64);
            s0 = wmma_bf16(a0, bfr, s0);
            v16bf a1  = load_frag_rm(&Ql[(tm1 * 16) * 64 + ks * 32], 64);
            s1 = wmma_bf16(a1, bfr, s1);
        }
#pragma unroll
        for (int r = 0; r < 8; ++r) {
            Sl[(tm0 * 16 + mlo + r) * 64 + nn] = s0[r] * 0.125f;  // 1/sqrt(64)
            Sl[(tm1 * 16 + mlo + r) * 64 + nn] = s1[r] * 0.125f;
        }
        __syncthreads();

        if (tid < 64) {
            int jmax = len - kb * 64; if (jmax > 64) jmax = 64;
            const float* srow = &Sl[tid * 64];
            float rmax = -3.0e38f;
            for (int j = 0; j < jmax; ++j) rmax = fmaxf(rmax, srow[j]);
            float m0 = mrow[tid];
            float nm = fmaxf(m0, rmax);
            float al = __expf(m0 - nm);
            float ps = 0.f;
            for (int j = 0; j < 64; ++j) {
                float p = (j < jmax) ? __expf(srow[j] - nm) : 0.f;
                Pl[tid * 64 + j] = (__bf16)p;
                ps += p;
            }
            lrow[tid] = lrow[tid] * al + ps;
            mrow[tid] = nm;
            arow[tid] = al;
        }
        __syncthreads();

#pragma unroll
        for (int r = 0; r < 8; ++r) {
            o0[r] *= arow[tm0 * 16 + mlo + r];
            o1[r] *= arow[tm1 * 16 + mlo + r];
        }
#pragma unroll
        for (int ks = 0; ks < 2; ++ks) {
            v16bf bfr = load_frag_rm(&VTl[(tn0 * 16) * 64 + ks * 32], 64);
            v16bf a0  = load_frag_rm(&Pl[(tm0 * 16) * 64 + ks * 32], 64);
            o0 = wmma_bf16(a0, bfr, o0);
            v16bf a1  = load_frag_rm(&Pl[(tm1 * 16) * 64 + ks * 32], 64);
            o1 = wmma_bf16(a1, bfr, o1);
        }
    }
    __syncthreads();
#pragma unroll
    for (int r = 0; r < 8; ++r) {
        int r0 = tm0 * 16 + mlo + r;
        float l0 = lrow[r0], i0 = (l0 > 0.f) ? 1.0f / l0 : 0.f;
        out[((size_t)(b * TT + qb * 64 + r0)) * DD + h * DKh + nn] = (__bf16)(o0[r] * i0);
        int r1 = tm1 * 16 + mlo + r;
        float l1 = lrow[r1], i1 = (l1 > 0.f) ? 1.0f / l1 : 0.f;
        out[((size_t)(b * TT + qb * 64 + r1)) * DD + h * DKh + nn] = (__bf16)(o1[r] * i1);
    }
}

// ---------------------------------------------------------------------------
// 128x128 tiled WMMA GEMM  C = A @ W^T + bias, TDM-staged double-buffered LDS.
// 8 waves, each owns a 64(M) x 32(N) sub-tile = eight 16x16 accumulators;
// B-fragments reused across the 4 M-subtiles.
// MODE 0 (gated residual): out32[row,n] = resid[row,n] + emb[b,gate+n]*C
// MODE 1 (gelu):           out16[row,n] = bf16(gelu_tanh(C))
// ---------------------------------------------------------------------------
template <int MODE>
__global__ void gemm128_kernel(const __bf16* __restrict__ A,
                               const __bf16* __restrict__ W,
                               const float* __restrict__ bias,
                               int Kd, int Mtot, int Ntot,
                               const float* __restrict__ resid,
                               const float* __restrict__ emb, int gate_off,
                               float* __restrict__ out32,
                               __bf16* __restrict__ out16) {
    __shared__ __bf16 Al[2][128 * 64];
    __shared__ __bf16 Bl[2][128 * 64];
    int n0 = blockIdx.x * 128;
    int row0 = blockIdx.y * 128;
    int tid = threadIdx.x, wid = tid >> 5, lane = tid & 31;
    int wm = wid >> 2, wn = wid & 3;            // 2 x 4 wave grid
    v8f acc[4][2] = {};

    // prologue: TDM-load first K-chunk (wave 0 issues, all waves wait at bar)
    if (wid == 0) {
        tdm_load_2d(A + (size_t)row0 * Kd, &Al[0][0], Kd, Mtot, 128, 64);
        tdm_load_2d(W + (size_t)n0 * Kd,  &Bl[0][0], Kd, Ntot, 128, 64);
        __builtin_amdgcn_s_wait_tensorcnt(0);
    }
    __syncthreads();

    int nk = Kd >> 6;
    for (int kc = 0; kc < nk; ++kc) {
        int buf = kc & 1;
        // prefetch next chunk into the other buffer while computing this one
        if (wid == 0 && kc + 1 < nk) {
            tdm_load_2d(A + (size_t)row0 * Kd + (kc + 1) * 64, &Al[buf ^ 1][0],
                        Kd, Mtot, 128, 64);
            tdm_load_2d(W + (size_t)n0 * Kd + (kc + 1) * 64, &Bl[buf ^ 1][0],
                        Kd, Ntot, 128, 64);
        }
#pragma unroll
        for (int ks = 0; ks < 2; ++ks) {
            v16bf bf0 = load_frag_rm(&Bl[buf][(wn * 32) * 64 + ks * 32], 64);
            v16bf bf1 = load_frag_rm(&Bl[buf][(wn * 32 + 16) * 64 + ks * 32], 64);
#pragma unroll
            for (int i = 0; i < 4; ++i) {
                v16bf af = load_frag_rm(&Al[buf][(wm * 64 + i * 16) * 64 + ks * 32], 64);
                acc[i][0] = wmma_bf16(af, bf0, acc[i][0]);
                acc[i][1] = wmma_bf16(af, bf1, acc[i][1]);
            }
        }
        if (wid == 0) __builtin_amdgcn_s_wait_tensorcnt(0);
        __syncthreads();
    }

    int mlo = (lane >> 4) * 8;
#pragma unroll
    for (int i = 0; i < 4; ++i) {
#pragma unroll
        for (int j = 0; j < 2; ++j) {
            int nn = n0 + wn * 32 + j * 16 + (lane & 15);
            float bo = bias[nn];
#pragma unroll
            for (int r = 0; r < 8; ++r) {
                int row = row0 + wm * 64 + i * 16 + mlo + r;
                float cv = acc[i][j][r] + bo;
                if (MODE == 0) {
                    int bi = row >> 10;   // T=1024 rows per batch
                    float g = emb[(size_t)bi * SIXD + gate_off + nn];
                    out32[(size_t)row * Ntot + nn] =
                        resid[(size_t)row * Ntot + nn] + g * cv;
                } else {
                    out16[(size_t)row * Ntot + nn] = (__bf16)gelu_tanh(cv);
                }
            }
        }
    }
}

// ---------------------------------------------------------------------------
// Host-side orchestration (all launches on `stream`; scratch carved from d_ws)
// ---------------------------------------------------------------------------
static inline char* carve(char*& p, size_t bytes) {
    char* r = p;
    p += (bytes + 255) & ~(size_t)255;
    return r;
}

extern "C" void kernel_launch(void* const* d_in, const int* in_sizes, int n_in,
                              void* d_out, int out_size, void* d_ws, size_t ws_size,
                              hipStream_t stream) {
    const float* noisy   = (const float*)d_in[0];
    const float* clean   = (const float*)d_in[1];
    const float* tvec    = (const float*)d_in[2];
    const float* ln_ng   = (const float*)d_in[3];
    const float* ln_nb   = (const float*)d_in[4];
    const float* ln_cg   = (const float*)d_in[5];
    const float* ln_cb   = (const float*)d_in[6];
    const float* ada_w   = (const float*)d_in[7];
    const float* ada_b   = (const float*)d_in[8];
    const float* wq      = (const float*)d_in[9];
    const float* bq      = (const float*)d_in[10];
    const float* wk      = (const float*)d_in[11];
    const float* bk      = (const float*)d_in[12];
    const float* wv      = (const float*)d_in[13];
    const float* bv      = (const float*)d_in[14];
    const float* fc_w    = (const float*)d_in[15];
    const float* fc_b    = (const float*)d_in[16];
    const float* ff_w1   = (const float*)d_in[17];
    const float* ff_b1   = (const float*)d_in[18];
    const float* ff_w2   = (const float*)d_in[19];
    const float* ff_b2   = (const float*)d_in[20];
    const int*   clen    = (const int*)d_in[22];
    float* outp          = (float*)d_out;

    char* p = (char*)d_ws;
    __bf16* st_pad   = (__bf16*)carve(p, (size_t)16 * DD * 2);
    float*  emb      = (float*) carve(p, (size_t)BB * SIXD * 4);
    float*  noisy_ln = (float*) carve(p, (size_t)BB * TT * DD * 4);
    __bf16* clean_bf = (__bf16*)carve(p, (size_t)BB * TT * DD * 2);
    __bf16* normx_bf = (__bf16*)carve(p, (size_t)BB * TT * DD * 2);
    __bf16* q_bf     = (__bf16*)carve(p, (size_t)BB * TT * DD * 2);
    __bf16* k_bf     = (__bf16*)carve(p, (size_t)BB * TT * DD * 2);
    __bf16* v_bf     = (__bf16*)carve(p, (size_t)BB * TT * DD * 2);
    __bf16* attn_bf  = (__bf16*)carve(p, (size_t)BB * TT * DD * 2);
    float*  x_f32    = (float*) carve(p, (size_t)BB * TT * DD * 4);
    __bf16* n2_bf    = (__bf16*)carve(p, (size_t)BB * TT * DD * 2);
    __bf16* h1_bf    = (__bf16*)carve(p, (size_t)BB * TT * FFd * 2);
    __bf16* ada_wbf  = (__bf16*)carve(p, (size_t)SIXD * DD * 2);
    __bf16* wq_bf    = (__bf16*)carve(p, (size_t)DD * DKh * KW * 2);
    __bf16* wk_bf    = (__bf16*)carve(p, (size_t)DD * DKh * KW * 2);
    __bf16* wv_bf    = (__bf16*)carve(p, (size_t)DD * DKh * KW * 2);
    __bf16* fc_wbf   = (__bf16*)carve(p, (size_t)DD * DD * 2);
    __bf16* ff1_wbf  = (__bf16*)carve(p, (size_t)FFd * DD * 2);
    __bf16* ff2_wbf  = (__bf16*)carve(p, (size_t)DD * FFd * 2);
    if ((size_t)(p - (char*)d_ws) > ws_size) return;  // scratch too small

    // 1) weight conversion to bf16
    f32_to_bf16_kernel<<<1024, 256, 0, stream>>>(ada_w, ada_wbf, (size_t)SIXD * DD);
    f32_to_bf16_kernel<<<512, 256, 0, stream>>>(wq, wq_bf, (size_t)DD * DKh * KW);
    f32_to_bf16_kernel<<<512, 256, 0, stream>>>(wk, wk_bf, (size_t)DD * DKh * KW);
    f32_to_bf16_kernel<<<512, 256, 0, stream>>>(wv, wv_bf, (size_t)DD * DKh * KW);
    f32_to_bf16_kernel<<<512, 256, 0, stream>>>(fc_w, fc_wbf, (size_t)DD * DD);
    f32_to_bf16_kernel<<<1024, 256, 0, stream>>>(ff_w1, ff1_wbf, (size_t)FFd * DD);
    f32_to_bf16_kernel<<<1024, 256, 0, stream>>>(ff_w2, ff2_wbf, (size_t)DD * FFd);

    // 2) pre-norms (eps 1e-5, affine)
    ln_rows_kernel<<<BB * TT, 256, 0, stream>>>(noisy, ln_ng, ln_nb, 1e-5f, noisy_ln, nullptr);
    ln_rows_kernel<<<BB * TT, 256, 0, stream>>>(clean, ln_cg, ln_cb, 1e-5f, nullptr, clean_bf);

    // 3) AdaLayerNormZero embedding: emb = silu(t) @ ada_w^T + ada_b
    silu_pad_kernel<<<64, 256, 0, stream>>>(tvec, st_pad);
    ada_gemm_kernel<<<SIXD / 16, 32, 0, stream>>>(st_pad, ada_wbf, ada_b, emb);

    // 4) normx = LN(noisy)*(1+scale_msa)+shift_msa   (split: shift@0, scale@D)
    ln_mod_kernel<<<BB * TT, 256, 0, stream>>>(noisy_ln, emb, 0, DD, normx_bf);

    // 5) grouped-conv QKV (WMMA)
    conv_qkv_kernel<<<BB * HH * (TT / 64), 256, 0, stream>>>(normx_bf, wq_bf, bq, q_bf);
    conv_qkv_kernel<<<BB * HH * (TT / 64), 256, 0, stream>>>(clean_bf, wk_bf, bk, k_bf);
    conv_qkv_kernel<<<BB * HH * (TT / 64), 256, 0, stream>>>(clean_bf, wv_bf, bv, v_bf);

    // 6) masked flash cross-attention
    attn_kernel<<<BB * HH * (TT / 64), 256, 0, stream>>>(q_bf, k_bf, v_bf, clen, attn_bf);

    // 7) x = noisy_ln + gate_msa * (attn @ fc_w^T + fc_b)   (gate_msa @ 2D)
    gemm128_kernel<0><<<dim3(DD / 128, BB * TT / 128), 256, 0, stream>>>(
        attn_bf, fc_wbf, fc_b, DD, BB * TT, DD, noisy_ln, emb, 2 * DD, x_f32, nullptr);

    // 8) n2 = LN(x)*(1+scale_mlp)+shift_mlp   (shift@3D, scale@4D)
    ln_mod_kernel<<<BB * TT, 256, 0, stream>>>(x_f32, emb, 3 * DD, 4 * DD, n2_bf);

    // 9) h1 = gelu(n2 @ ff_w1^T + b1)
    gemm128_kernel<1><<<dim3(FFd / 128, BB * TT / 128), 256, 0, stream>>>(
        n2_bf, ff1_wbf, ff_b1, DD, BB * TT, FFd, nullptr, nullptr, 0, nullptr, h1_bf);

    // 10) out = x + gate_mlp * (h1 @ ff_w2^T + b2)   (gate_mlp @ 5D)
    gemm128_kernel<0><<<dim3(DD / 128, BB * TT / 128), 256, 0, stream>>>(
        h1_bf, ff2_wbf, ff_b2, FFd, BB * TT, DD, x_f32, emb, 5 * DD, outp, nullptr);
}
